// Model_1047972020547
// MI455X (gfx1250) — compile-verified, matrix operand order checked by reading, not run on hardware
//
#include <hip/hip_runtime.h>
#include <math.h>

// ---------------------------------------------------------------------------
// Problem constants (from reference)
// ---------------------------------------------------------------------------
#define V_   50000
#define E_   300
#define EP_  320      // E padded to multiple of 32 (bf16 WMMA K-step)
#define H_   256
#define G3_  768      // 3*H
#define R_   2048
#define L0_  128
#define LR_  64
#define S_   2049     // sequence length (1 + R)
#define SP_  2064     // S padded to multiple of 16 (WMMA M tile)

typedef __attribute__((ext_vector_type(16))) __bf16 v16bf;
typedef __attribute__((ext_vector_type(8)))  __bf16 v8bf;
typedef __attribute__((ext_vector_type(8)))  float  v8f;

__device__ __forceinline__ v16bf cat8(v8bf lo, v8bf hi) {
  return __builtin_shufflevector(lo, hi, 0,1,2,3,4,5,6,7,8,9,10,11,12,13,14,15);
}

// ---------------------------------------------------------------------------
// Workspace layout (bytes, 256B-aligned blocks). Total ~17.4 MB.
// ---------------------------------------------------------------------------
constexpr size_t AL(size_t x) { return (x + 255) & ~(size_t)255; }
constexpr size_t SZ_XBF  = (size_t)SP_ * EP_ * 2;      // X bf16, padded
constexpr size_t SZ_WIH  = (size_t)G3_ * EP_ * 2;      // Wih bf16, K-padded
constexpr size_t SZ_WL   = (size_t)H_ * (2*H_) * 2;    // Wl bf16
constexpr size_t SZ_GI   = (size_t)SP_ * G3_ * 4;      // Gi fp32
constexpr size_t SZ_OUT2 = (size_t)SP_ * (2*H_) * 2;   // out2 bf16
constexpr size_t OFF_XBF  = 0;
constexpr size_t OFF_WIHF = AL(OFF_XBF  + SZ_XBF);
constexpr size_t OFF_WIHB = AL(OFF_WIHF + SZ_WIH);
constexpr size_t OFF_WL   = AL(OFF_WIHB + SZ_WIH);
constexpr size_t OFF_GIF  = AL(OFF_WL   + SZ_WL);
constexpr size_t OFF_GIB  = AL(OFF_GIF  + SZ_GI);
constexpr size_t OFF_OUT2 = AL(OFF_GIB  + SZ_GI);
constexpr size_t OFF_HF   = AL(OFF_OUT2 + SZ_OUT2);
constexpr size_t OFF_HB   = AL(OFF_HF   + H_*4);
constexpr size_t OFF_SC   = AL(OFF_HB   + H_*4);

// ---------------------------------------------------------------------------
// Kernel 1: embedding gather+sum -> X rows, stored bf16 with K-padding.
// Row 0 = sum of L0 original-token embeddings; rows 1..2048 = masked reply
// sums. Table (60 MB) is L2-resident; accesses over `e` are coalesced.
// ---------------------------------------------------------------------------
__global__ __launch_bounds__(EP_) void embed_rows_kernel(
    const int* __restrict__ orig_tok, const int* __restrict__ rep_tok,
    const int* __restrict__ rep_len,  const float* __restrict__ embed,
    __bf16* __restrict__ Xbf) {
  const int row = blockIdx.x;     // 0..SP_-1
  const int e   = threadIdx.x;    // 0..EP_-1
  float s = 0.0f;
  if (row < S_ && e < E_) {
    if (row == 0) {
      for (int t = 0; t < L0_; ++t)
        s += embed[(size_t)orig_tok[t] * E_ + e];
    } else {
      const int r = row - 1;
      const int L = rep_len[r];
      const int* tk = rep_tok + (size_t)r * LR_;
      for (int t = 0; t < L; ++t)
        s += embed[(size_t)tk[t] * E_ + e];
    }
  }
  Xbf[(size_t)row * EP_ + e] = (__bf16)s;   // pad cols/rows get exact zeros
}

// ---------------------------------------------------------------------------
// Kernel 2: fp32 -> bf16 weight conversion with K padding (zeros).
// ---------------------------------------------------------------------------
__global__ void f32_to_bf16_pad(const float* __restrict__ src,
                                __bf16* __restrict__ dst,
                                int rows, int cols, int ldp) {
  const int total = rows * ldp;
  for (int idx = blockIdx.x * blockDim.x + threadIdx.x; idx < total;
       idx += gridDim.x * blockDim.x) {
    const int r = idx / ldp, c = idx - r * ldp;
    dst[idx] = (c < cols) ? (__bf16)src[(size_t)r * cols + c] : (__bf16)0.0f;
  }
}

// ---------------------------------------------------------------------------
// Kernel 3: Gi = X @ Wih^T for both directions via v_wmma_f32_16x16x32_bf16.
// One wave = one 16x16 output tile; K = 320 -> 10 WMMA steps.
// A fragment (16x32 bf16, ISA 7.12.2): lanes 0-15 hold K [k0..k0+7, k0+16..23],
// lanes 16-31 hold K [k0+8..15, k0+24..31]; B mirrored (N = lane&15), which
// maps to contiguous reads of Wih rows (B[k][n] = Wih[n][k]).
// ---------------------------------------------------------------------------
__global__ __launch_bounds__(256) void gemm_gi_kernel(
    const __bf16* __restrict__ Xbf, const __bf16* __restrict__ Wf,
    const __bf16* __restrict__ Wb, float* __restrict__ Gif,
    float* __restrict__ Gib) {
  const int MT = SP_ / 16, NT = G3_ / 16;       // 129, 48
  const int w    = blockIdx.x * 8 + (threadIdx.x >> 5);
  const int lane = threadIdx.x & 31;
  const int dir  = w / (MT * NT);
  const int rem  = w - dir * (MT * NT);
  const int mt   = rem / NT, nt = rem - mt * NT;
  const __bf16* Wmat = dir ? Wb : Wf;
  float* G = dir ? Gib : Gif;

  const int arow = mt * 16 + (lane & 15);
  const int bcol = nt * 16 + (lane & 15);
  const int khi  = (lane >> 4) << 3;            // +8 elems for lanes 16-31
  const __bf16* ap = Xbf  + (size_t)arow * EP_ + khi;
  const __bf16* bp = Wmat + (size_t)bcol * EP_ + khi;

  v8f c = {};
  #pragma unroll
  for (int kt = 0; kt < EP_ / 32; ++kt) {
    v16bf a = cat8(*(const v8bf*)ap, *(const v8bf*)(ap + 16));
    v16bf b = cat8(*(const v8bf*)bp, *(const v8bf*)(bp + 16));
    c = __builtin_amdgcn_wmma_f32_16x16x32_bf16(false, a, false, b,
                                                (short)0, c, false, false);
    ap += 32; bp += 32;
  }
  const int rbase = mt * 16 + ((lane >> 4) << 3);   // M = v + 8*(lane>=16)
  const int col   = nt * 16 + (lane & 15);
  #pragma unroll
  for (int v = 0; v < 8; ++v)
    G[(size_t)(rbase + v) * G3_ + col] = c[v];
}

// ---------------------------------------------------------------------------
// Kernel 4: the sequential GRU recurrence (both directions as 2 blocks).
// h in LDS; each of 256 threads owns 3 Whh rows (L2-resident, 786 KB).
// Writes out2 (bf16, fwd cols 0..255 / bwd cols 256..511) + final h.
// ---------------------------------------------------------------------------
__global__ __launch_bounds__(H_) void gru_recurrence_kernel(
    const float* __restrict__ Gif, const float* __restrict__ Gib,
    const float* __restrict__ Whh_f, const float* __restrict__ bih_f,
    const float* __restrict__ bhh_f, const float* __restrict__ Whh_b,
    const float* __restrict__ bih_b, const float* __restrict__ bhh_b,
    __bf16* __restrict__ out2, float* __restrict__ h_f,
    float* __restrict__ h_b) {
  const int dir = blockIdx.x, j = threadIdx.x;
  const float* Gi  = dir ? Gib   : Gif;
  const float* Whh = dir ? Whh_b : Whh_f;
  const float* bih = dir ? bih_b : bih_f;
  const float* bhh = dir ? bhh_b : bhh_f;
  float* hout = dir ? h_b : h_f;
  const int cbase = dir ? H_ : 0;

  __shared__ float h[H_];
  h[j] = 0.0f;
  const float bi_r = bih[j], bi_z = bih[H_ + j], bi_n = bih[2*H_ + j];
  const float bh_r = bhh[j], bh_z = bhh[H_ + j], bh_n = bhh[2*H_ + j];
  const float* wr = Whh + (size_t)j * H_;
  const float* wz = Whh + (size_t)(H_ + j) * H_;
  const float* wn = Whh + (size_t)(2*H_ + j) * H_;
  __syncthreads();

  for (int t = 0; t < S_; ++t) {
    const int row = dir ? (S_ - 1 - t) : t;
    const float* g = Gi + (size_t)row * G3_;
    const int nrow = dir ? (row > 0 ? row - 1 : 0)
                         : (row + 1 < S_ ? row + 1 : row);
    __builtin_prefetch(Gi + (size_t)nrow * G3_ + j, 0, 0);  // global_prefetch_b8

    float ar = bh_r, az = bh_z, an = bh_n;
    #pragma unroll 4
    for (int k = 0; k < H_; ++k) {
      const float hk = h[k];
      ar = fmaf(wr[k], hk, ar);
      az = fmaf(wz[k], hk, az);
      an = fmaf(wn[k], hk, an);
    }
    const float hold = h[j];
    const float r = 1.0f / (1.0f + expf(-(g[j]         + bi_r + ar)));
    const float z = 1.0f / (1.0f + expf(-(g[H_ + j]    + bi_z + az)));
    const float n = tanhf(g[2*H_ + j] + bi_n + r * an);
    const float hnew = fmaf(z, hold - n, n);   // (1-z)*n + z*hold
    __syncthreads();                 // all reads of h done
    h[j] = hnew;
    out2[(size_t)row * (2*H_) + cbase + j] = (__bf16)hnew;
    __syncthreads();                 // new h visible
  }
  hout[j] = h[j];
}

// ---------------------------------------------------------------------------
// Kernel 5: output = out2 @ Wl^T + bl (WMMA bf16, K = 512 -> 16 steps),
// stored fp32 straight into d_out (rows < S_ guarded).
// ---------------------------------------------------------------------------
__global__ __launch_bounds__(256) void gemm_out_kernel(
    const __bf16* __restrict__ out2, const __bf16* __restrict__ Wlbf,
    const float* __restrict__ bl, float* __restrict__ out) {
  const int MT = SP_ / 16, NT = H_ / 16;        // 129, 16
  const int w    = blockIdx.x * 8 + (threadIdx.x >> 5);
  const int lane = threadIdx.x & 31;
  const int mt   = w / NT, nt = w - mt * NT;
  const int K    = 2 * H_;                       // 512

  const int arow = mt * 16 + (lane & 15);
  const int bcol = nt * 16 + (lane & 15);
  const int khi  = (lane >> 4) << 3;
  const __bf16* ap = out2 + (size_t)arow * K + khi;
  const __bf16* bp = Wlbf + (size_t)bcol * K + khi;

  v8f c = {};
  #pragma unroll
  for (int kt = 0; kt < K / 32; ++kt) {
    v16bf a = cat8(*(const v8bf*)ap, *(const v8bf*)(ap + 16));
    v16bf b = cat8(*(const v8bf*)bp, *(const v8bf*)(bp + 16));
    c = __builtin_amdgcn_wmma_f32_16x16x32_bf16(false, a, false, b,
                                                (short)0, c, false, false);
    ap += 32; bp += 32;
  }
  const int rbase = mt * 16 + ((lane >> 4) << 3);
  const int col   = nt * 16 + (lane & 15);
  const float bias = bl[col];
  #pragma unroll
  for (int v = 0; v < 8; ++v) {
    const int r = rbase + v;
    if (r < S_) out[(size_t)r * H_ + col] = c[v] + bias;
  }
}

// ---------------------------------------------------------------------------
// Kernel 6: attention + sigmoid + loss (single 1024-thread block).
// scores = output @ (h_f+h_b); softmax; ctx = w @ output; result; loss.
// ---------------------------------------------------------------------------
__global__ __launch_bounds__(1024) void attention_kernel(
    const float* __restrict__ out, const float* __restrict__ h_f,
    const float* __restrict__ h_b, const float* __restrict__ Wo,
    const float* __restrict__ bo, const float* __restrict__ label,
    float* __restrict__ scores, float* __restrict__ loss_res) {
  const int tid = threadIdx.x;
  __shared__ float hsum[H_];
  __shared__ float red[1024];
  if (tid < H_) hsum[tid] = h_f[tid] + h_b[tid];
  __syncthreads();

  float lmax = -3.0e38f;
  for (int t = tid; t < S_; t += 1024) {
    const float* o = out + (size_t)t * H_;
    float s = 0.0f;
    for (int k = 0; k < H_; ++k) s = fmaf(o[k], hsum[k], s);
    scores[t] = s;
    lmax = fmaxf(lmax, s);
  }
  red[tid] = lmax; __syncthreads();
  for (int off = 512; off > 0; off >>= 1) {
    if (tid < off) red[tid] = fmaxf(red[tid], red[tid + off]);
    __syncthreads();
  }
  const float m = red[0]; __syncthreads();

  float lsum = 0.0f;
  for (int t = tid; t < S_; t += 1024) {
    const float e = expf(scores[t] - m);
    scores[t] = e;
    lsum += e;
  }
  red[tid] = lsum; __syncthreads();
  for (int off = 512; off > 0; off >>= 1) {
    if (tid < off) red[tid] += red[tid + off];
    __syncthreads();
  }
  const float inv = 1.0f / red[0];
  __syncthreads();

  if (tid < H_) {
    float c = 0.0f;
    for (int t = 0; t < S_; ++t)
      c = fmaf(scores[t], out[(size_t)t * H_ + tid], c);
    red[tid] = c * inv * Wo[tid];
  }
  __syncthreads();
  for (int off = H_ / 2; off > 0; off >>= 1) {
    if (tid < off) red[tid] += red[tid + off];
    __syncthreads();
  }
  if (tid == 0) {
    const float res = 1.0f / (1.0f + expf(-(red[0] + bo[0])));
    const float d = label[0] - res;
    loss_res[0] = d * d;     // loss
    loss_res[1] = res;       // result
  }
}

// ---------------------------------------------------------------------------
// Host-side launch sequence (graph-capture safe: launches only).
// ---------------------------------------------------------------------------
extern "C" void kernel_launch(void* const* d_in, const int* in_sizes, int n_in,
                              void* d_out, int out_size, void* d_ws,
                              size_t ws_size, hipStream_t stream) {
  const int*   orig_tok = (const int*)d_in[0];
  const int*   rep_tok  = (const int*)d_in[1];
  const int*   rep_len  = (const int*)d_in[2];
  const float* label    = (const float*)d_in[3];
  const float* embed    = (const float*)d_in[4];
  const float* Wih_f    = (const float*)d_in[5];
  const float* Whh_f    = (const float*)d_in[6];
  const float* bih_f    = (const float*)d_in[7];
  const float* bhh_f    = (const float*)d_in[8];
  const float* Wih_b    = (const float*)d_in[9];
  const float* Whh_b    = (const float*)d_in[10];
  const float* bih_b    = (const float*)d_in[11];
  const float* bhh_b    = (const float*)d_in[12];
  const float* Wl       = (const float*)d_in[13];
  const float* bl       = (const float*)d_in[14];
  const float* Wo       = (const float*)d_in[15];
  const float* bo       = (const float*)d_in[16];

  float* out = (float*)d_out;                 // [S_*H_] output, then loss, result
  char*  ws  = (char*)d_ws;
  __bf16* Xbf    = (__bf16*)(ws + OFF_XBF);
  __bf16* Wihbf_f= (__bf16*)(ws + OFF_WIHF);
  __bf16* Wihbf_b= (__bf16*)(ws + OFF_WIHB);
  __bf16* Wlbf   = (__bf16*)(ws + OFF_WL);
  float*  Gif    = (float*)(ws + OFF_GIF);
  float*  Gib    = (float*)(ws + OFF_GIB);
  __bf16* out2   = (__bf16*)(ws + OFF_OUT2);
  float*  hF     = (float*)(ws + OFF_HF);
  float*  hB     = (float*)(ws + OFF_HB);
  float*  scores = (float*)(ws + OFF_SC);

  // 1) build X (bf16, padded)
  embed_rows_kernel<<<SP_, EP_, 0, stream>>>(orig_tok, rep_tok, rep_len, embed, Xbf);
  // 2) weights -> bf16 (padded K)
  f32_to_bf16_pad<<<480, 256, 0, stream>>>(Wih_f, Wihbf_f, G3_, E_, EP_);
  f32_to_bf16_pad<<<480, 256, 0, stream>>>(Wih_b, Wihbf_b, G3_, E_, EP_);
  f32_to_bf16_pad<<<256, 256, 0, stream>>>(Wl, Wlbf, H_, 2 * H_, 2 * H_);
  // 3) Gi = X @ Wih^T for both dirs (WMMA):  2*129*48 tiles / 8 waves per block
  gemm_gi_kernel<<<(2 * (SP_/16) * (G3_/16)) / 8, 256, 0, stream>>>(
      Xbf, Wihbf_f, Wihbf_b, Gif, Gib);
  // 4) sequential recurrence, 2 blocks (fwd/bwd)
  gru_recurrence_kernel<<<2, H_, 0, stream>>>(Gif, Gib, Whh_f, bih_f, bhh_f,
                                              Whh_b, bih_b, bhh_b, out2, hF, hB);
  // 5) output = out2 @ Wl^T + bl (WMMA) -> d_out
  gemm_out_kernel<<<((SP_/16) * (H_/16)) / 8, 256, 0, stream>>>(out2, Wlbf, bl, out);
  // 6) attention + loss -> d_out tail
  attention_kernel<<<1, 1024, 0, stream>>>(out, hF, hB, Wo, bo, label, scores,
                                           out + (size_t)S_ * H_);
}
